// MultiheadSelfAttention_11321533792930
// MI455X (gfx1250) — matrix operand order repaired and checked by reading, not compiled, not run on hardware
//
#include <hip/hip_runtime.h>
#include <cstdint>

// ---------------------------------------------------------------------------
// Multi-head self-attention forward for MI455X (gfx1250, wave32, WMMA).
// Matmuls run on v_wmma_f32_16x16x32_f16 (fp32 accum). The 512MB attn tensor
// is produced once, write-only: scores live in WGP LDS (128KB of 320KB), and
// the attn tile is pushed to HBM with GLOBAL_STORE_ASYNC_FROM_LDS_B128
// (ASYNCcnt path, 16B/lane, no VGPR round-trip).
// ---------------------------------------------------------------------------

typedef _Float16 v16h __attribute__((ext_vector_type(16)));
typedef float    v8f  __attribute__((ext_vector_type(8)));
typedef int      v4i  __attribute__((ext_vector_type(4)));

constexpr int B_  = 4;
constexpr int S_  = 2048;
constexpr int D_  = 512;
constexpr int H_  = 8;
constexpr int HD_ = 64;
constexpr float SCALE_ = 0.125f;      // 1/sqrt(64)

// LDS layout of attention kernel (floats)
constexpr int SC_FLOATS   = 16 * S_;              // 16 query rows x 2048 keys
constexpr int STAT_OFF    = SC_FLOATS;            // pmax[128] psum[128] rmax[16] rsum[16]
constexpr int PART_OFF    = STAT_OFF + 288;       // 4 waves x 16x64 partial ctx
constexpr int SMEM_FLOATS = PART_OFF + 4 * 16 * HD_;

#if defined(__has_builtin)
#if __has_builtin(__builtin_amdgcn_global_store_async_from_lds_b128) && \
    __has_builtin(__builtin_amdgcn_s_wait_asynccnt)
#define USE_ASYNC_LDS_STORE 1
#endif
#endif
#ifndef USE_ASYNC_LDS_STORE
#define USE_ASYNC_LDS_STORE 0
#endif

static __device__ inline v8f wmma16(v16h a, v16h b, v8f c) {
  // (neg_a, A, neg_b, B, c_mod, C, reuse_a, reuse_b)
  return __builtin_amdgcn_wmma_f32_16x16x32_f16(false, a, false, b, (short)0, c,
                                                false, false);
}

// A operand (16x32, MxK): lanes 0-15 -> M=lane, K={0..7,16..23};
// lanes 16-31 -> M=lane-16, K={8..15,24..31}. (ISA 7.12.2)
static __device__ inline v16h load_a_f32(const float* base, int lda, int mbase,
                                         int kbase, int lane) {
  const int r    = lane & 15;
  const int koff = (lane < 16) ? 0 : 8;
  const float* p = base + (size_t)(mbase + r) * lda + kbase + koff;
  v16h a;
#pragma unroll
  for (int i = 0; i < 8; ++i) { a[i] = (_Float16)p[i]; a[i + 8] = (_Float16)p[i + 16]; }
  return a;
}

static __device__ inline v16h load_a_f16(const _Float16* base, int lda, int mbase,
                                         int kbase, int lane) {
  const int r    = lane & 15;
  const int koff = (lane < 16) ? 0 : 8;
  const _Float16* p = base + (size_t)(mbase + r) * lda + kbase + koff;
  v16h a;
#pragma unroll
  for (int i = 0; i < 8; ++i) { a[i] = p[i]; a[i + 8] = p[i + 16]; }
  return a;
}

// B operand (32x16, KxN): lane holds column N=lane&15; lanes 0-15 carry
// K={0..7,16..23}, lanes 16-31 carry K={8..15,24..31} (matches A).
static __device__ inline v16h load_b_f32(const float* base, int ldb, int nbase,
                                         int kbase, int lane) {
  const int n  = lane & 15;
  const int kh = (lane < 16) ? 0 : 8;
  const float* p = base + (size_t)(nbase + n) * ldb + kbase + kh;
  v16h b;
#pragma unroll
  for (int i = 0; i < 8; ++i) { b[i] = (_Float16)p[i]; b[i + 8] = (_Float16)p[i + 16]; }
  return b;
}

static __device__ inline v16h load_b_f16(const _Float16* base, int ldb, int nbase,
                                         int kbase, int lane) {
  const int n  = lane & 15;
  const int kh = (lane < 16) ? 0 : 8;
  const _Float16* p = base + (size_t)(nbase + n) * ldb + kbase + kh;
  v16h b;
#pragma unroll
  for (int i = 0; i < 8; ++i) { b[i] = p[i]; b[i + 8] = p[i + 16]; }
  return b;
}

// ---------------------------------------------------------------------------
// GEMM: C[M,512] = A[M,512] @ W[512,512]^T + bias     (x @ W.T + b)
// MODE 0: store f16 row-major        (Q/K projections)
// MODE 1: store f16 V-transposed     -> Vt[B,H,HD,S]  (V projection)
// MODE 2: store f32 row-major        (output projection)
// Each wave computes a 16x64 tile (4 WMMA accumulators), 16 K-steps of 32.
// ---------------------------------------------------------------------------
template <int MODE, bool AF16>
__global__ __launch_bounds__(128) void gemm_wmma_k(const void* __restrict__ A,
                                                   const float* __restrict__ W,
                                                   const float* __restrict__ bias,
                                                   void* __restrict__ Cout) {
  const int lane  = threadIdx.x & 31;
  const int wave  = threadIdx.x >> 5;
  const int wid   = blockIdx.x * 4 + wave;   // 4096 wave tiles
  const int mbase = (wid >> 3) * 16;         // 512 m-tiles
  const int nb    = (wid & 7) * 64;          // 8 n-blocks of 64

  v8f acc[4] = {};
#pragma unroll 1
  for (int kb = 0; kb < D_; kb += 32) {
    v16h a;
    if (AF16) a = load_a_f16((const _Float16*)A, D_, mbase, kb, lane);
    else      a = load_a_f32((const float*)A,    D_, mbase, kb, lane);
#pragma unroll
    for (int t = 0; t < 4; ++t) {
      v16h bm = load_b_f32(W, D_, nb + t * 16, kb, lane);   // row n of W = col n of W.T
      acc[t] = wmma16(a, bm, acc[t]);
    }
  }

  const int n    = lane & 15;
  const int moff = (lane < 16) ? 0 : 8;
#pragma unroll
  for (int t = 0; t < 4; ++t) {
    const int col  = nb + t * 16 + n;
    const float bv = bias[col];
#pragma unroll
    for (int j = 0; j < 8; ++j) {
      const int m   = mbase + moff + j;
      const float v = acc[t][j] + bv;
      if (MODE == 2) {
        ((float*)Cout)[(size_t)m * D_ + col] = v;
      } else if (MODE == 0) {
        ((_Float16*)Cout)[(size_t)m * D_ + col] = (_Float16)v;
      } else {  // MODE 1: Vt[b][h][hd][s]
        const int bb = m >> 11, s = m & (S_ - 1);
        const int hh = col >> 6, hd = col & 63;
        ((_Float16*)Cout)[(((size_t)bb * H_ + hh) * HD_ + hd) * S_ + s] = (_Float16)v;
      }
    }
  }
}

// ---------------------------------------------------------------------------
// Attention: one block (4 waves) per (b, h, 16-query tile).
// Phase 1: scores = Q K^T * scale into LDS (each wave owns 512 keys).
// Phase 2: row softmax in LDS (8 threads x float4 per query row), then the
//          16x2048 attn tile goes to HBM — async LDS->global when available.
// Phase 3: ctx = P @ V via WMMA (Vt layout gives contiguous B-operand loads),
//          cross-wave reduction in LDS, f16 ctx stored for the Wo GEMM.
// ---------------------------------------------------------------------------
__global__ __launch_bounds__(128) void attn_wmma_k(const _Float16* __restrict__ Q,
                                                   const _Float16* __restrict__ Kmat,
                                                   const _Float16* __restrict__ Vt,
                                                   float* __restrict__ attn,
                                                   _Float16* __restrict__ ctx) {
  extern __shared__ float smem[];
  const int tid  = threadIdx.x;
  const int lane = tid & 31;
  const int wave = tid >> 5;

  const int qt = blockIdx.x & (S_ / 16 - 1);     // 128 query tiles
  const int h  = (blockIdx.x >> 7) & (H_ - 1);
  const int b  = blockIdx.x >> 10;
  const int qbase = qt * 16;

  const _Float16* Qp = Q    + ((size_t)b * S_ + qbase) * D_ + h * HD_;
  const _Float16* Kp = Kmat + (size_t)b * S_ * D_ + h * HD_;
  const _Float16* Vp = Vt   + (size_t)(b * H_ + h) * HD_ * S_;
  float* attp = attn + ((size_t)(b * H_ + h) * S_ + qbase) * S_;

  // Q tile A-operands for the two 32-wide K steps of HD=64
  const v16h aq0 = load_a_f16(Qp, D_, 0, 0, lane);
  const v16h aq1 = load_a_f16(Qp, D_, 0, 32, lane);

  // ---- Phase 1: raw scores into LDS ----
#pragma unroll 1
  for (int t = 0; t < 32; ++t) {
    const int keyb = wave * 512 + t * 16;
    const v16h b0 = load_b_f16(Kp, D_, keyb, 0, lane);
    const v16h b1 = load_b_f16(Kp, D_, keyb, 32, lane);
    v8f sc = {};
    sc = wmma16(aq0, b0, sc);
    sc = wmma16(aq1, b1, sc);
    const int n = lane & 15, moff = (lane < 16) ? 0 : 8;
#pragma unroll
    for (int j = 0; j < 8; ++j)
      smem[(size_t)(moff + j) * S_ + keyb + n] = sc[j] * SCALE_;
  }
  __syncthreads();

  // ---- Phase 2: softmax over each 2048-key row (float4, v_max_num path) ----
  float* pmax = smem + STAT_OFF;    // [128]
  float* psum = pmax + 128;         // [128]
  float* rmax = psum + 128;         // [16]
  float* rsum = rmax + 16;          // [16]
  const int row = tid >> 3, seg = tid & 7;
  float4* srow4 = (float4*)(smem + (size_t)row * S_ + seg * 256);

  float mx = -3.4e38f;
#pragma unroll 1
  for (int i = 0; i < 64; ++i) {
    const float4 v = srow4[i];
    mx = __builtin_fmaxf(mx, __builtin_fmaxf(__builtin_fmaxf(v.x, v.y),
                                             __builtin_fmaxf(v.z, v.w)));
  }
  pmax[tid] = mx;
  __syncthreads();
  if (seg == 0) {
    float m2 = pmax[row * 8];
#pragma unroll
    for (int i = 1; i < 8; ++i) m2 = __builtin_fmaxf(m2, pmax[row * 8 + i]);
    rmax[row] = m2;
  }
  __syncthreads();
  const float rm = rmax[row];
  float sum = 0.f;
#pragma unroll 1
  for (int i = 0; i < 64; ++i) {
    float4 v = srow4[i];
    v.x = __expf(v.x - rm);
    v.y = __expf(v.y - rm);
    v.z = __expf(v.z - rm);
    v.w = __expf(v.w - rm);
    srow4[i] = v;
    sum += (v.x + v.y) + (v.z + v.w);
  }
  psum[tid] = sum;
  __syncthreads();
  if (seg == 0) {
    float s2 = 0.f;
#pragma unroll
    for (int i = 0; i < 8; ++i) s2 += psum[row * 8 + i];
    rsum[row] = s2;
  }
  __syncthreads();
  const float inv = 1.0f / rsum[row];
#pragma unroll 1
  for (int i = 0; i < 64; ++i) {
    float4 v = srow4[i];
    v.x *= inv; v.y *= inv; v.z *= inv; v.w *= inv;
    srow4[i] = v;
  }
  __syncthreads();

  // ---- Write attn tile to HBM exactly once (write-only) ----
#if USE_ASYNC_LDS_STORE
  {
    // 16B per lane per issue, LDS -> global with no VGPR data movement.
    // Builtin params are pointers to 16B int vectors in AS1 (global) / AS3 (LDS).
    typedef __attribute__((address_space(1))) v4i gv4_t;
    typedef __attribute__((address_space(3))) v4i lv4_t;
#pragma unroll 1
    for (int i = tid; i < (16 * S_) / 4; i += 128) {
      __builtin_amdgcn_global_store_async_from_lds_b128(
          (gv4_t*)(attp + 4 * (size_t)i), (lv4_t*)(smem + 4 * (size_t)i), 0, 0);
    }
  }
#else
  {
    const float4* s4 = (const float4*)smem;
    float4* a4 = (float4*)attp;                 // 16 contiguous rows of S floats
#pragma unroll 1
    for (int i = tid; i < (16 * S_) / 4; i += 128) a4[i] = s4[i];
  }
#endif

  // ---- Phase 3: ctx = P @ V (each wave: its 512 keys, full 16x64 tile) ----
  // (reads the same LDS region the async stores read; no one overwrites it)
  v8f oacc[4] = {};
#pragma unroll 1
  for (int t = 0; t < 16; ++t) {
    const int kb = wave * 512 + t * 32;
    const int r = lane & 15;
    const int koff = (lane < 16) ? 0 : 8;
    const float* pp = smem + (size_t)r * S_ + kb + koff;
    v16h pa;
#pragma unroll
    for (int i = 0; i < 8; ++i) { pa[i] = (_Float16)pp[i]; pa[i + 8] = (_Float16)pp[i + 16]; }
#pragma unroll
    for (int nt = 0; nt < 4; ++nt) {
      const v16h vb = load_b_f16(Vp, S_, nt * 16, kb, lane);  // Vt rows = hd, contiguous keys
      oacc[nt] = wmma16(pa, vb, oacc[nt]);
    }
  }
  // partial ctx per wave -> LDS, reduce across 4 waves, store f16 ctx
  float* part = smem + PART_OFF;
  {
    const int n = lane & 15, moff = (lane < 16) ? 0 : 8;
#pragma unroll
    for (int nt = 0; nt < 4; ++nt)
#pragma unroll
      for (int j = 0; j < 8; ++j)
        part[wave * (16 * HD_) + (moff + j) * HD_ + nt * 16 + n] = oacc[nt][j];
  }
  __syncthreads();
#pragma unroll 1
  for (int idx = tid; idx < 16 * HD_; idx += 128) {
    const float v = part[idx] + part[idx + 16 * HD_] + part[idx + 2 * 16 * HD_] +
                    part[idx + 3 * 16 * HD_];
    const int q = idx >> 6, hd = idx & 63;
    ctx[((size_t)b * S_ + qbase + q) * D_ + h * HD_ + hd] = (_Float16)v;
  }
#if USE_ASYNC_LDS_STORE
  __builtin_amdgcn_s_wait_asynccnt(0);  // attn stores complete before retire
#endif
}

// ---------------------------------------------------------------------------
extern "C" void kernel_launch(void* const* d_in, const int* in_sizes, int n_in,
                              void* d_out, int out_size, void* d_ws, size_t ws_size,
                              hipStream_t stream) {
  (void)in_sizes; (void)n_in; (void)out_size; (void)ws_size;
  const float* query = (const float*)d_in[0];
  const float* key_  = (const float*)d_in[1];
  const float* value = (const float*)d_in[2];
  const float* Wq = (const float*)d_in[3];
  const float* bq = (const float*)d_in[4];
  const float* Wk = (const float*)d_in[5];
  const float* bk = (const float*)d_in[6];
  const float* Wv = (const float*)d_in[7];
  const float* bv = (const float*)d_in[8];
  const float* Wo = (const float*)d_in[9];
  const float* bo = (const float*)d_in[10];

  float* out  = (float*)d_out;
  float* attn = out + (size_t)B_ * S_ * D_;   // outputs concatenated: out, attn

  char* ws = (char*)d_ws;
  const size_t HALF = (size_t)B_ * S_ * D_ * sizeof(_Float16);  // 8 MB
  _Float16* q16 = (_Float16*)(ws);
  _Float16* k16 = (_Float16*)(ws + HALF);
  _Float16* vT  = (_Float16*)(ws + 2 * HALF);
  _Float16* ctx = (_Float16*)(ws + 3 * HALF);

  // 4096 wave tiles of 16x64 over [8192 x 512], 4 waves/block
  gemm_wmma_k<0, false><<<1024, 128, 0, stream>>>(query, Wq, bq, q16);
  gemm_wmma_k<0, false><<<1024, 128, 0, stream>>>(key_,  Wk, bk, k16);
  gemm_wmma_k<1, false><<<1024, 128, 0, stream>>>(value, Wv, bv, vT);

  const size_t smem_bytes = (size_t)SMEM_FLOATS * sizeof(float);  // ~145 KB < 320 KB WGP LDS
  attn_wmma_k<<<B_ * H_ * (S_ / 16), 128, smem_bytes, stream>>>(q16, k16, vT, attn, ctx);

  gemm_wmma_k<2, true><<<1024, 128, 0, stream>>>(ctx, Wo, bo, out);
}
